// NlmCNN_7172595385039
// MI455X (gfx1250) — compile-verified, matrix operand order checked by reading, not run on hardware
//
#include <hip/hip_runtime.h>
#include <hip/hip_bf16.h>

typedef __attribute__((ext_vector_type(16))) _Float16 v16h;
typedef __attribute__((ext_vector_type(8)))  float    v8f;

#define HP 258              // padded spatial (256 + 1-px halo each side)
#define CH 64               // hidden channels
#define KK 13               // dynamic filter size
#define HO 244              // output spatial (256 - 13 + 1)

// workspace layout (bytes)
#define WS_W2_OFF   0                       // 72 frags * 512 f16  = 73728 B
#define WS_W3_OFF   73728                   // 198 frags * 512 f16 = 202752 B
#define WS_H1_OFF   276480                  // 8*258*258*64 f16 = 68161536 B
#define WS_H2_OFF   (276480 + 68161536)

// ---------------------------------------------------------------------------
// Kernel 0: repack W2 (64x576) and W3 (169->176 x 576) into WMMA A-fragment
// order. Fragment = (mtile, ktile); per lane L: M = mt*16 + (L&15),
// K slots i=0..15 -> kk = (i/8)*16 + (L>=16)*8 + (i%8)  (ISA 16-bit A 16x32
// layout). k ordering: k = tap*64 + ci  (tap = u*3+v).
// ---------------------------------------------------------------------------
__global__ void pack_weights(const float* __restrict__ W2,
                             const float* __restrict__ W3,
                             _Float16* __restrict__ fW2,
                             _Float16* __restrict__ fW3) {
    const int NW2 = 72 * 512;   // 4 mtiles * 18 ktiles
    const int NW3 = 198 * 512;  // 11 mtiles * 18 ktiles
    int tid = blockIdx.x * blockDim.x + threadIdx.x;
    if (tid >= NW2 + NW3) return;
    bool isW3 = (tid >= NW2);
    int e    = isW3 ? tid - NW2 : tid;
    int frag = e >> 9;
    int r    = e & 511;
    int lane = r >> 4;
    int i    = r & 15;
    int mt   = frag / 18;
    int kt   = frag % 18;
    int kk   = ((i >> 3) << 4) + ((lane >> 4) << 3) + (i & 7);
    int k    = kt * 32 + kk;
    int tap  = k >> 6;          // u*3+v
    int ci   = k & 63;          // input channel
    int M    = mt * 16 + (lane & 15);
    float val;
    if (isW3) val = (M < 169) ? W3[(M * 64 + ci) * 9 + tap] : 0.0f;
    else      val = W2[(M * 64 + ci) * 9 + tap];
    (isW3 ? fW3 : fW2)[(size_t)frag * 512 + r] = (_Float16)val;
}

// ---------------------------------------------------------------------------
// Kernel 1: conv1 (1->64, 3x3 SAME) + bias + ReLU -> h1 padded NHWC f16.
// Also zero-fills the 1-px halo of BOTH h1 and h2 (conv2/conv3 read halos).
// ---------------------------------------------------------------------------
__global__ void conv1_relu(const float* __restrict__ x,
                           const float* __restrict__ W1,
                           const float* __restrict__ b1,
                           _Float16* __restrict__ h1p,
                           _Float16* __restrict__ h2p) {
    __shared__ float sW[CH * 9];
    __shared__ float sB[CH];
    for (int t = threadIdx.x; t < CH * 9; t += blockDim.x) sW[t] = W1[t];
    if (threadIdx.x < CH) sB[threadIdx.x] = b1[threadIdx.x];
    __syncthreads();

    int tid = blockIdx.x * blockDim.x + threadIdx.x;
    const int total = 8 * HP * HP;
    if (tid >= total) return;
    int b  = tid / (HP * HP);
    int rp = tid % (HP * HP);
    int yp = rp / HP, xp = rp % HP;

    __attribute__((aligned(16))) _Float16 out[CH];
    bool border = (yp == 0) | (yp == HP - 1) | (xp == 0) | (xp == HP - 1);
    if (border) {
        #pragma unroll
        for (int c = 0; c < CH; ++c) out[c] = (_Float16)0.0f;
    } else {
        int y = yp - 1, xc = xp - 1;
        float xv[9];
        #pragma unroll
        for (int du = 0; du < 3; ++du)
            #pragma unroll
            for (int dv = 0; dv < 3; ++dv) {
                int yy = y + du - 1, xx = xc + dv - 1;
                bool ok = (yy >= 0) & (yy < 256) & (xx >= 0) & (xx < 256);
                xv[du * 3 + dv] = ok ? x[((size_t)b * 256 + yy) * 256 + xx] : 0.0f;
            }
        for (int c = 0; c < CH; ++c) {
            float s = sB[c];
            #pragma unroll
            for (int t = 0; t < 9; ++t) s = fmaf(sW[c * 9 + t], xv[t], s);
            out[c] = (_Float16)fmaxf(s, 0.0f);
        }
    }
    size_t base = (size_t)tid * CH;
    #pragma unroll
    for (int q = 0; q < 8; ++q)
        ((uint4*)(h1p + base))[q] = ((const uint4*)out)[q];
    if (border) {
        #pragma unroll
        for (int q = 0; q < 8; ++q)
            ((uint4*)(h2p + base))[q] = ((const uint4*)out)[q];
    }
}

// B-fragment load helper: lane n<16 = pixel n, K 0..7 & 16..23;
// lane n>=16 = pixel n-16, K 8..15 & 24..31 (mirrors A layout).
__device__ inline v16h load_bfrag(const _Float16* p) {
    v16h bf;
    ((uint4*)&bf)[0] = *(const uint4*)(p);
    ((uint4*)&bf)[1] = *(const uint4*)(p + 16);
    return bf;
}

// ---------------------------------------------------------------------------
// Kernel 2: conv2 (64->64) implicit GEMM on v_wmma_f32_16x16x32_f16.
// Block = 256 threads = 8 waves; wave = 32 pixels x 64 out-ch
// (4 M-tiles x 2 N-tiles) so each A-fragment feeds 2 WMMAs.
// K loop fully unrolled -> all loads are immediate offsets off 2 base ptrs.
// Grid = 8 batches * 256 rows (block covers a full row).
// ---------------------------------------------------------------------------
__global__ void conv2_wmma(const _Float16* __restrict__ h1p,
                           const _Float16* __restrict__ fW2,
                           const float* __restrict__ b2,
                           _Float16* __restrict__ h2p) {
    int wv = threadIdx.x >> 5, lane = threadIdx.x & 31;
    int bid = blockIdx.x;
    int b   = bid >> 8;
    int y   = bid & 255;
    int x0  = wv * 32;
    int pix = lane & 15;
    int hi  = lane >> 4;

    const _Float16* bbase0 = h1p +
        (((size_t)(b * HP + y) * HP) + (x0 + pix)) * CH + hi * 8;
    const _Float16* bbase1 = bbase0 + 16 * CH;
    const _Float16* abase  = fW2 + lane * 16;

    v8f acc[4][2] = {};
    #pragma unroll
    for (int tap = 0; tap < 9; ++tap) {
        const int u = tap / 3, v = tap % 3;
        #pragma unroll
        for (int half = 0; half < 2; ++half) {
            const int kt  = tap * 2 + half;
            const int off = (u * HP + v) * CH + half * 32;
            v16h bf0 = load_bfrag(bbase0 + off);
            v16h bf1 = load_bfrag(bbase1 + off);
            #pragma unroll
            for (int mt = 0; mt < 4; ++mt) {
                v16h af = *(const v16h*)(abase + (mt * 18 + kt) * 512);
                acc[mt][0] = __builtin_amdgcn_wmma_f32_16x16x32_f16(
                    false, af, false, bf0, (short)0, acc[mt][0], false, false);
                acc[mt][1] = __builtin_amdgcn_wmma_f32_16x16x32_f16(
                    false, af, false, bf1, (short)0, acc[mt][1], false, false);
            }
        }
    }
    // epilogue: bias + ReLU + cvt f16, NHWC store (padded coords)
    #pragma unroll
    for (int mt = 0; mt < 4; ++mt) {
        int chb = mt * 16 + hi * 8;
        #pragma unroll
        for (int nt = 0; nt < 2; ++nt) {
            union { _Float16 h[8]; uint4 u4; } o;
            #pragma unroll
            for (int j = 0; j < 8; ++j)
                o.h[j] = (_Float16)fmaxf(acc[mt][nt][j] + b2[chb + j], 0.0f);
            _Float16* dst = h2p +
                (((size_t)(b * HP + (y + 1)) * HP) + (x0 + nt * 16 + pix + 1)) * CH + chb;
            *(uint4*)dst = o.u4;
        }
    }
}

// ---------------------------------------------------------------------------
// Kernel 3: conv3 (64->169, computed only on the consumed 244x244 region)
// fused with the dynamic-filter patch dot.  Block = 64 threads = 2 waves,
// wave = 32 pixels x 176 ch (11 M-tiles x 2 N-tiles, M padded 169->176).
// w stays in registers -> LDS -> consumed; never written to HBM.
// Per-lane pixel coordinates clamped ONCE so all tap offsets are immediates.
// Grid = 8 * 244 * 4 quads of 64 pixels.
// ---------------------------------------------------------------------------
__global__ void conv3_fused(const _Float16* __restrict__ h2p,
                            const _Float16* __restrict__ fW3,
                            const float* __restrict__ b3,
                            const float* __restrict__ x,
                            float* __restrict__ yout) {
    __shared__ float ldsW[2][176 * 32];   // [ch][pixel-in-wave]
    __shared__ float ldsX[2][13 * 44];    // 13 rows x 44 cols strip
    int wv = threadIdx.x >> 5, lane = threadIdx.x & 31;
    int bid  = blockIdx.x;
    int b    = bid / (HO * 4);
    int rem  = bid % (HO * 4);
    int i    = rem >> 2;          // output row
    int quad = rem & 3;
    int j0   = quad * 64 + wv * 32;
    int pix  = lane & 15;
    int hi   = lane >> 4;
    int r    = i + 6;             // w-space row

    // clamp dead lanes (j >= 244) to a safe in-bounds column, once per N-tile
    int jp0 = min(j0 + pix,      HO - 1);
    int jp1 = min(j0 + 16 + pix, HO - 1);
    const _Float16* bbase0 = h2p +
        (((size_t)(b * HP + r) * HP) + (jp0 + 6)) * CH + hi * 8;
    const _Float16* bbase1 = h2p +
        (((size_t)(b * HP + r) * HP) + (jp1 + 6)) * CH + hi * 8;
    const _Float16* abase = fW3 + lane * 16;

    v8f acc[11][2] = {};
    #pragma unroll
    for (int tap = 0; tap < 9; ++tap) {
        const int u = tap / 3, v = tap % 3;
        #pragma unroll
        for (int half = 0; half < 2; ++half) {
            const int kt  = tap * 2 + half;
            const int off = (u * HP + v) * CH + half * 32;
            v16h bf0 = load_bfrag(bbase0 + off);
            v16h bf1 = load_bfrag(bbase1 + off);
            #pragma unroll
            for (int mt = 0; mt < 11; ++mt) {
                v16h af = *(const v16h*)(abase + (mt * 18 + kt) * 512);
                acc[mt][0] = __builtin_amdgcn_wmma_f32_16x16x32_f16(
                    false, af, false, bf0, (short)0, acc[mt][0], false, false);
                acc[mt][1] = __builtin_amdgcn_wmma_f32_16x16x32_f16(
                    false, af, false, bf1, (short)0, acc[mt][1], false, false);
            }
        }
    }
    // dump dynamic filters (+bias) to LDS: ldsW[ch][pixel]
    #pragma unroll
    for (int mt = 0; mt < 11; ++mt) {
        int chb = mt * 16 + hi * 8;
        #pragma unroll
        for (int j = 0; j < 8; ++j) {
            int ch = chb + j;
            float bias = (ch < 169) ? b3[ch] : 0.0f;
            ldsW[wv][ch * 32 + pix]      = acc[mt][0][j] + bias;
            ldsW[wv][ch * 32 + 16 + pix] = acc[mt][1][j] + bias;
        }
    }
    // stage the x patch strip: rows i..i+12, cols j0..j0+43
    for (int t = lane; t < 13 * 44; t += 32) {
        int rr = t / 44, cc = t % 44;
        int colx = min(j0 + cc, 255);
        ldsX[wv][t] = x[((size_t)(b * 256) + (i + rr)) * 256 + colx];
    }
    __syncthreads();
    // per-pixel 169-term dot: one pixel per lane, lanes with j >= 244 idle
    int j = j0 + lane;
    if (j < HO) {
        float s = 0.0f;
        #pragma unroll
        for (int u = 0; u < KK; ++u)
            #pragma unroll
            for (int v = 0; v < KK; ++v)
                s = fmaf(ldsW[wv][(u * KK + v) * 32 + lane],
                         ldsX[wv][u * 44 + lane + v], s);
        yout[((size_t)(b * HO) + i) * HO + j] = s;
    }
}

// ---------------------------------------------------------------------------
extern "C" void kernel_launch(void* const* d_in, const int* in_sizes, int n_in,
                              void* d_out, int out_size, void* d_ws, size_t ws_size,
                              hipStream_t stream) {
    const float* x  = (const float*)d_in[0];
    const float* W1 = (const float*)d_in[1];
    const float* b1 = (const float*)d_in[2];
    const float* W2 = (const float*)d_in[3];
    const float* b2 = (const float*)d_in[4];
    const float* W3 = (const float*)d_in[5];
    const float* b3 = (const float*)d_in[6];
    float* y = (float*)d_out;

    char* ws = (char*)d_ws;
    _Float16* fW2 = (_Float16*)(ws + WS_W2_OFF);
    _Float16* fW3 = (_Float16*)(ws + WS_W3_OFF);
    _Float16* h1p = (_Float16*)(ws + WS_H1_OFF);
    _Float16* h2p = (_Float16*)(ws + WS_H2_OFF);

    // 0) pack weights into WMMA A-fragment order
    {
        int n = (72 + 198) * 512;
        pack_weights<<<(n + 255) / 256, 256, 0, stream>>>(W2, W3, fW2, fW3);
    }
    // 1) conv1 + ReLU -> h1 (and zero halos of h1/h2)
    {
        int n = 8 * HP * HP;
        conv1_relu<<<(n + 255) / 256, 256, 0, stream>>>(x, W1, b1, h1p, h2p);
    }
    // 2) conv2 WMMA + ReLU -> h2
    conv2_wmma<<<8 * 256, 256, 0, stream>>>(h1p, fW2, b2, h2p);
    // 3) conv3 WMMA fused with patch dot -> y
    conv3_fused<<<8 * HO * 4, 64, 0, stream>>>(h2p, fW3, b3, x, y);
}
